// sLSTMCell_69698729279878
// MI455X (gfx1250) — matrix operand order repaired and checked by reading, not compiled, not run on hardware
//
#include <hip/hip_runtime.h>

#define Bsz   8192
#define Ddim  1024
#define Hh    8
#define HDdim 128

typedef __attribute__((ext_vector_type(16))) __bf16 v16bf;
typedef __attribute__((ext_vector_type(8)))  float  v8f;
typedef __attribute__((ext_vector_type(4)))  float  f32x4;

__device__ inline __bf16 f2bf(float f) {
    union { float f; unsigned u; } a; a.f = f;
    unsigned r = a.u + 0x7FFFu + ((a.u >> 16) & 1u);   // round-to-nearest-even
    unsigned short h = (unsigned short)(r >> 16);
    return __builtin_bit_cast(__bf16, h);
}

__device__ inline float fast_rcp(float x)   { return __builtin_amdgcn_rcpf(x); }
__device__ inline float fast_sigmoid(float x) { return fast_rcp(1.f + __expf(-x)); }
// tanh(x) = 1 - 2/(exp(2x)+1)
__device__ inline float fast_tanh(float x)  { return 1.f - 2.f * fast_rcp(__expf(2.f * x) + 1.f); }
// log(sigmoid(x)) = min(x,0) - log(1 + exp(-|x|))
__device__ inline float fast_logsig(float x) {
    return fminf(x, 0.f) - __logf(1.f + __expf(-fabsf(x)));
}

// ---------------------------------------------------------------------------
// Kernel 0: zero the all_zero(n) flag
// ---------------------------------------------------------------------------
__global__ void k_zero_flag(int* flag) {
    if (threadIdx.x == 0 && blockIdx.x == 0) *flag = 0;
}

// ---------------------------------------------------------------------------
// Kernel 1: flag = (any element of n nonzero)
// ---------------------------------------------------------------------------
__global__ void k_reduce_n(const f32x4* __restrict__ n4, int* __restrict__ flag) {
    __shared__ int s;
    if (threadIdx.x == 0) s = 0;
    __syncthreads();
    bool any = false;
    size_t base = (size_t)blockIdx.x * 1024 + threadIdx.x;
    #pragma unroll
    for (int j = 0; j < 4; ++j) {
        f32x4 v = __builtin_nontemporal_load(&n4[base + (size_t)j * 256]);
        any |= (v.x != 0.f) | (v.y != 0.f) | (v.z != 0.f) | (v.w != 0.f);
    }
    if (any) s = 1;
    __syncthreads();
    if (threadIdx.x == 0 && s) atomicOr(flag, 1);
}

// ---------------------------------------------------------------------------
// Kernel 2: pre-swizzle Wi/Wf/Wz/Wo (f32) into bf16 B-matrix fragments.
// B fragment k order chosen here (contiguous, same as A) for V_WMMA_F32_16X16X32_BF16 (32x16, KxN):
//   lanes 0..15 : N = lane,     K = kt*32 + 0..15   (elements 0..15)
//   lanes 16..31: N = lane-16,  K = kt*32 + 16..31
// Each frag = 32 lanes x 16 bf16 = 1024 B; lane's 16 values contiguous (32 B).
// fragId = ((g*8 + head)*8 + nt)*4 + kt
// ---------------------------------------------------------------------------
__global__ void k_swizzle_w(const float* __restrict__ Wi, const float* __restrict__ Wf,
                            const float* __restrict__ Wz, const float* __restrict__ Wo,
                            __bf16* __restrict__ wsw) {
    int gid = blockIdx.x * 256 + threadIdx.x;      // 524288 total
    int j      = gid & 15;
    int lane   = (gid >> 4) & 31;
    int fragId = gid >> 9;
    int kt   = fragId & 3;
    int nt   = (fragId >> 2) & 7;
    int head = (fragId >> 5) & 7;
    int g    = fragId >> 8;
    const float* W = (g == 0) ? Wi : (g == 1) ? Wf : (g == 2) ? Wz : Wo;
    int n = nt * 16 + (lane & 15);
    int k = kt * 32 + ((lane >> 4) << 4) + j;
    wsw[gid] = f2bf(W[((size_t)head * HDdim + k) * HDdim + n]);
}

// ---------------------------------------------------------------------------
// Kernel 3: main fused sLSTM cell.
// Block = 256 threads (8 waves), tile = 32 rows x 1 head (128 cols).
// Wave (wr,wc): rows wr*16..+15, cols wc*32..+31 (2 N-tiles of 16).
// ---------------------------------------------------------------------------
__global__ __launch_bounds__(256)
void k_slstm(const float* __restrict__ i_in, const float* __restrict__ f_in,
             const float* __restrict__ z_in, const float* __restrict__ o_in,
             const float* __restrict__ c_in, const float* __restrict__ m_in,
             const float* __restrict__ h_in, const float* __restrict__ n_st,
             const float* __restrict__ bi,  const float* __restrict__ bfg,
             const float* __restrict__ bz,  const float* __restrict__ bo,
             const __bf16* __restrict__ wsw, const int* __restrict__ flag,
             float* __restrict__ out)
{
    // 272-byte row stride: 16B-aligned, rotates LDS banks by 4/row -> no conflicts
    __shared__ __align__(16) __bf16 hs[32][136];

    const int tid     = threadIdx.x;
    const int head    = blockIdx.y;
    const int rowBase = blockIdx.x * 32;

    // ---- stage h tile (32 rows x 128 cols) as bf16 into LDS, coalesced NT ----
    const f32x4* h4 = (const f32x4*)h_in;   // row = 256 float4
    #pragma unroll
    for (int it = 0; it < 4; ++it) {
        int slot = tid + it * 256;            // 0..1023
        int r  = slot >> 5;
        int c4 = slot & 31;
        f32x4 v = __builtin_nontemporal_load(&h4[(size_t)(rowBase + r) * 256 + head * 32 + c4]);
        __bf16* dst = &hs[r][c4 * 4];
        dst[0] = f2bf(v.x); dst[1] = f2bf(v.y); dst[2] = f2bf(v.z); dst[3] = f2bf(v.w);
    }
    __syncthreads();

    const int lane = tid & 31;
    const int wave = tid >> 5;
    const int wr   = wave >> 2;               // 0..1
    const int wc   = wave & 3;                // 0..3
    const int mrow = wr * 16 + (lane & 15);
    const int hi   = lane >> 4;               // 0 or 1

    // ---- A fragments (16x32 bf16) for the 4 K-steps, kept in VGPRs ----
    // lanes 0..15: K = kt*32 + 0..15; lanes 16..31: K = kt*32 + 16..31
    v16bf afr[4];
    #pragma unroll
    for (int kt = 0; kt < 4; ++kt) {
        int k0 = kt * 32 + hi * 16;
        union { v16bf v; uint4 q[2]; } u;
        u.q[0] = *(const uint4*)&hs[mrow][k0];
        u.q[1] = *(const uint4*)&hs[mrow][k0 + 8];
        afr[kt] = u.v;
    }

    // ---- 4 gates x 2 N-tiles x 4 K-steps of bf16 WMMA, f32 accumulate ----
    v8f acc[4][2];
    #pragma unroll
    for (int g = 0; g < 4; ++g) {
        #pragma unroll
        for (int t = 0; t < 2; ++t) {
            int nt = wc * 2 + t;
            v8f c = {0.f, 0.f, 0.f, 0.f, 0.f, 0.f, 0.f, 0.f};
            #pragma unroll
            for (int kt = 0; kt < 4; ++kt) {
                int fragId = ((g * 8 + head) * 8 + nt) * 4 + kt;
                union { v16bf v; uint4 q[2]; } ub;
                const uint4* p = (const uint4*)(wsw + (size_t)fragId * 512 + lane * 16);
                ub.q[0] = p[0];
                ub.q[1] = p[1];
                c = __builtin_amdgcn_wmma_f32_16x16x32_bf16(
                        false, afr[kt], false, ub.v, (short)0, c, false, false);
            }
            acc[g][t] = c;
        }
    }

    const bool allZero = (*flag == 0);
    const size_t BD = (size_t)Bsz * Ddim;

    // ---- epilogue: C layout is VGPR v -> M = v + 8*hi, lane%16 -> N ----
    #pragma unroll
    for (int t = 0; t < 2; ++t) {
        int d = head * 128 + wc * 32 + t * 16 + (lane & 15);
        float vbi = bi[d], vbf = bfg[d], vbz = bz[d], vbo = bo[d];
        #pragma unroll
        for (int v = 0; v < 8; ++v) {
            int row = rowBase + wr * 16 + v + hi * 8;
            size_t idx = (size_t)row * Ddim + d;

            float iv = __builtin_nontemporal_load(&i_in[idx]) + acc[0][t][v] + vbi;
            float fv = __builtin_nontemporal_load(&f_in[idx]) + acc[1][t][v] + vbf;
            float zv = __builtin_nontemporal_load(&z_in[idx]) + acc[2][t][v] + vbz;
            float ov = __builtin_nontemporal_load(&o_in[idx]) + acc[3][t][v] + vbo;

            float osig = fast_sigmoid(ov);
            float logf = fast_logsig(fv);
            float mo   = __builtin_nontemporal_load(&m_in[idx]);
            float mn   = allZero ? iv : fmaxf(logf + mo, iv);
            float ip   = fminf(__expf(iv - mn), 1.f);
            float fp   = fminf(__expf(logf + mo - mn), 1.f);
            float cn   = fp * __builtin_nontemporal_load(&c_in[idx]) + ip * fast_tanh(zv);
            float nn   = fp * __builtin_nontemporal_load(&n_st[idx]) + ip;
            float ht   = cn * fast_rcp(fmaxf(nn, 1e-6f));
            float hn   = osig * ht;

            __builtin_nontemporal_store(cn, &out[idx]);
            __builtin_nontemporal_store(nn, &out[BD + idx]);
            __builtin_nontemporal_store(mn, &out[2 * BD + idx]);
            __builtin_nontemporal_store(hn, &out[3 * BD + idx]);
        }
    }
}

// ---------------------------------------------------------------------------
extern "C" void kernel_launch(void* const* d_in, const int* in_sizes, int n_in,
                              void* d_out, int out_size, void* d_ws, size_t ws_size,
                              hipStream_t stream) {
    // setup_inputs() dict order: i f z o c m h n Wi Wf Wz Wo bi bf bz bo
    const float* i_in = (const float*)d_in[0];
    const float* f_in = (const float*)d_in[1];
    const float* z_in = (const float*)d_in[2];
    const float* o_in = (const float*)d_in[3];
    const float* c_in = (const float*)d_in[4];
    const float* m_in = (const float*)d_in[5];
    const float* h_in = (const float*)d_in[6];
    const float* n_st = (const float*)d_in[7];
    const float* Wi   = (const float*)d_in[8];
    const float* Wf   = (const float*)d_in[9];
    const float* Wz   = (const float*)d_in[10];
    const float* Wo   = (const float*)d_in[11];
    const float* bi   = (const float*)d_in[12];
    const float* bfg  = (const float*)d_in[13];
    const float* bz   = (const float*)d_in[14];
    const float* bo   = (const float*)d_in[15];
    float* out = (float*)d_out;

    int*    flag = (int*)d_ws;
    __bf16* wsw  = (__bf16*)((char*)d_ws + 256);

    k_zero_flag<<<1, 32, 0, stream>>>(flag);
    k_reduce_n<<<2048, 256, 0, stream>>>((const f32x4*)n_st, flag);
    k_swizzle_w<<<2048, 256, 0, stream>>>(Wi, Wf, Wz, Wo, wsw);

    dim3 grid(Bsz / 32, Hh);
    k_slstm<<<grid, 256, 0, stream>>>(i_in, f_in, z_in, o_in, c_in, m_in, h_in, n_st,
                                      bi, bfg, bz, bo, wsw, flag, out);
}
